// PartBasedGraphCNN_14663018348629
// MI455X (gfx1250) — compile-verified, matrix operand order checked by reading, not run on hardware
//
#include <hip/hip_runtime.h>

typedef __attribute__((ext_vector_type(16))) __bf16 v16bf;
typedef __attribute__((ext_vector_type(8)))  __bf16 v8bf;
typedef __attribute__((ext_vector_type(8)))  float  v8f;
typedef __attribute__((ext_vector_type(4)))  float  f32x4;

#define BATCH   32
#define HW      128
#define PDIM    64
#define NNODES  4096          // 64*64
#define CMID    32
#define CGCN    64
#define KFC1    262144        // 64 * 4096
#define OFC1    128
#define NCLS    101
#define SPLITS  256
#define KSTEPS  32            // (KFC1/32) / SPLITS k-steps of 32 per split

__device__ __forceinline__ v16bf cat8(v8bf lo, v8bf hi) {
  return __builtin_shufflevector(lo, hi, 0,1,2,3,4,5,6,7,8,9,10,11,12,13,14,15);
}

// ---------------------------------------------------------------------------
// Kernel 1: conv3x3 (SAME) + bias + relu + maxpool2x2  ->  pooled [B,32,64,64]
// relu(max(conv)+b) == max(relu(conv+b)) by monotonicity.
// ---------------------------------------------------------------------------
__global__ void conv_relu_pool(const float* __restrict__ x,
                               const float* __restrict__ cw,
                               const float* __restrict__ cb,
                               float* __restrict__ pooled) {
  __shared__ float wsm[27];
  int blk = blockIdx.x;
  int t  = blk & 15;          // 16 blocks per (b,oc) image
  int oc = (blk >> 4) & 31;
  int b  = blk >> 9;
  if (threadIdx.x < 27) wsm[threadIdx.x] = cw[oc * 27 + threadIdx.x];
  __syncthreads();

  int p  = t * 256 + threadIdx.x;      // 0..4095 pooled pixel
  int ph = p >> 6, pw = p & 63;
  const float* xb = x + (size_t)b * 3 * HW * HW;

  float m = -3.4e38f;
  #pragma unroll
  for (int dy = 0; dy < 2; ++dy) {
    #pragma unroll
    for (int dx = 0; dx < 2; ++dx) {
      int h = 2 * ph + dy, w = 2 * pw + dx;
      float s = 0.f;
      #pragma unroll
      for (int ic = 0; ic < 3; ++ic) {
        #pragma unroll
        for (int kh = 0; kh < 3; ++kh) {
          int hh = h + kh - 1;
          if (hh < 0 || hh >= HW) continue;
          #pragma unroll
          for (int kw = 0; kw < 3; ++kw) {
            int ww = w + kw - 1;
            if (ww < 0 || ww >= HW) continue;
            s += xb[(ic * HW + hh) * HW + ww] * wsm[(ic * 3 + kh) * 3 + kw];
          }
        }
      }
      m = fmaxf(m, s);
    }
  }
  pooled[((b * CMID + oc) * PDIM + ph) * PDIM + pw] = fmaxf(m + cb[oc], 0.f);
}

// ---------------------------------------------------------------------------
// Kernel 2: GCN linear via WMMA bf16: hfeat[131072,64] = xg[131072,32] @ gw[32,64]
// xg = pooled buffer raw-reinterpreted (faithful torch .view).
// One WMMA (K=32) per 16x16 output tile. 8 waves/block -> 32 rows x 64 cols.
// ---------------------------------------------------------------------------
__global__ void gcn_linear_wmma(const float* __restrict__ xg,
                                const float* __restrict__ gw,
                                float* __restrict__ hfeat) {
  int wave = threadIdx.x >> 5;
  int lane = threadIdx.x & 31;
  int rtile = blockIdx.x * 2 + (wave >> 2);
  int ctile = wave & 3;
  int group = lane >> 4;
  int row   = rtile * 16 + (lane & 15);
  int col   = ctile * 16 + (lane & 15);

  // A fragment: two contiguous K-runs [8g..8g+7], [16+8g..16+8g+7]
  const float* ap = xg + (size_t)row * 32 + group * 8;
  f32x4 a0 = *(const f32x4*)(ap);
  f32x4 a1 = *(const f32x4*)(ap + 4);
  f32x4 a2 = *(const f32x4*)(ap + 16);
  f32x4 a3 = *(const f32x4*)(ap + 20);
  v16bf A;
  #pragma unroll
  for (int j = 0; j < 4; ++j) {
    A[j]      = (__bf16)a0[j];
    A[4 + j]  = (__bf16)a1[j];
    A[8 + j]  = (__bf16)a2[j];
    A[12 + j] = (__bf16)a3[j];
  }
  // B fragment: same K mapping, N = lane%16, gw is [32,64] row-major (tiny, cached)
  v16bf Bf;
  #pragma unroll
  for (int j = 0; j < 8; ++j) {
    Bf[j]     = (__bf16)gw[(group * 8 + j) * 64 + col];
    Bf[8 + j] = (__bf16)gw[(16 + group * 8 + j) * 64 + col];
  }

  v8f c = {};
  c = __builtin_amdgcn_wmma_f32_16x16x32_bf16(false, A, false, Bf, (short)0, c,
                                              false, false);
  #pragma unroll
  for (int v = 0; v < 8; ++v)
    hfeat[(size_t)(rtile * 16 + v + 8 * group) * 64 + col] = c[v];
}

// ---------------------------------------------------------------------------
// Kernel 3: GCN sym-norm 5-point aggregation + bias + relu -> g (bf16)
// deg = #neighbors + self; coeff = rsqrt(deg_src)*rsqrt(deg_dst).
// hfeat (33.5 MB) is L2-resident, so the 5x reuse is cheap.
// ---------------------------------------------------------------------------
__global__ void gcn_agg(const float* __restrict__ hfeat,
                        const float* __restrict__ gb,
                        __bf16* __restrict__ g) {
  int idx  = blockIdx.x * 256 + threadIdx.x;     // 32*4096*64 = 8.39M
  int f    = idx & 63;
  int node = (idx >> 6) & 4095;
  int b    = idx >> 18;
  int h = node >> 6, w = node & 63;

  auto dinv = [](int hh, int ww) {
    int nb = (hh > 0) + (hh < 63) + (ww > 0) + (ww < 63);
    return rsqrtf((float)(nb + 1));
  };

  const float* hf = hfeat + ((size_t)b * NNODES) * 64 + f;
  float dd = dinv(h, w);
  float s = dd * dd * hf[node * 64];                         // self loop
  if (h > 0)  s += dd * dinv(h - 1, w) * hf[(node - 64) * 64];
  if (h < 63) s += dd * dinv(h + 1, w) * hf[(node + 64) * 64];
  if (w > 0)  s += dd * dinv(h, w - 1) * hf[(node - 1) * 64];
  if (w < 63) s += dd * dinv(h, w + 1) * hf[(node + 1) * 64];

  float r = fmaxf(s + gb[f], 0.f);
  g[((size_t)b * NNODES + node) * 64 + f] = (__bf16)r;
}

// ---------------------------------------------------------------------------
// Kernel 4: fc1 split-K GEMM via WMMA bf16.
// f1_partial[s, 32, 128]:  A = g_bf16 [32, 262144], B = fc1_w^T (row o contiguous in k).
// Grid = SPLITS * (2 mtiles * 8 ntiles), 1 wave per block; each wave streams a
// 64 KB slice (shared by both mtile waves) of the 134 MB weight matrix -> bandwidth-bound by design.
// ---------------------------------------------------------------------------
__global__ void fc1_gemm_wmma(const __bf16* __restrict__ g,
                              const float* __restrict__ w,
                              float* __restrict__ partials) {
  int lane   = threadIdx.x & 31;
  int ksplit = blockIdx.x >> 4;
  int tile   = blockIdx.x & 15;
  int mtile  = tile >> 3;            // 0..1
  int ntile  = tile & 7;             // 0..7
  int group  = lane >> 4;
  int mrow   = mtile * 16 + (lane & 15);   // batch row 0..31
  int ncol   = ntile * 16 + (lane & 15);   // out col 0..127

  const __bf16* ga = g + (size_t)mrow * KFC1;
  const float*  wb = w + (size_t)ncol * KFC1;

  v8f acc = {};
  int kbase = ksplit * (KSTEPS * 32);
  for (int s = 0; s < KSTEPS; ++s) {
    int k = kbase + s * 32;
    // A: bf16, two 16B vector loads
    v8bf alo = *(const v8bf*)(ga + k + group * 8);
    v8bf ahi = *(const v8bf*)(ga + k + 16 + group * 8);
    v16bf A = cat8(alo, ahi);
    // B: f32, four b128 loads + cvt to bf16
    const float* wp = wb + k + group * 8;
    f32x4 b0 = *(const f32x4*)(wp);
    f32x4 b1 = *(const f32x4*)(wp + 4);
    f32x4 b2 = *(const f32x4*)(wp + 16);
    f32x4 b3 = *(const f32x4*)(wp + 20);
    v16bf Bf;
    #pragma unroll
    for (int j = 0; j < 4; ++j) {
      Bf[j]      = (__bf16)b0[j];
      Bf[4 + j]  = (__bf16)b1[j];
      Bf[8 + j]  = (__bf16)b2[j];
      Bf[12 + j] = (__bf16)b3[j];
    }
    acc = __builtin_amdgcn_wmma_f32_16x16x32_bf16(false, A, false, Bf, (short)0,
                                                  acc, false, false);
  }

  float* out = partials + (size_t)ksplit * BATCH * OFC1;
  #pragma unroll
  for (int v = 0; v < 8; ++v)
    out[(mtile * 16 + v + 8 * group) * OFC1 + ncol] = acc[v];
}

// ---------------------------------------------------------------------------
// Kernel 5: deterministic fixed-order split-K reduction + bias + relu
// ---------------------------------------------------------------------------
__global__ void fc1_reduce(const float* __restrict__ partials,
                           const float* __restrict__ bias,
                           float* __restrict__ f1) {
  int i = blockIdx.x * 256 + threadIdx.x;   // 0..4095
  float s = 0.f;
  for (int sp = 0; sp < SPLITS; ++sp) s += partials[(size_t)sp * BATCH * OFC1 + i];
  f1[i] = fmaxf(s + bias[i & (OFC1 - 1)], 0.f);
}

// ---------------------------------------------------------------------------
// Kernel 6: fc2 (tiny: 32x101, K=128) in f32 for final precision
// ---------------------------------------------------------------------------
__global__ void fc2_kernel(const float* __restrict__ f1,
                           const float* __restrict__ w,
                           const float* __restrict__ bias,
                           float* __restrict__ out) {
  int i = blockIdx.x * 256 + threadIdx.x;
  if (i >= BATCH * NCLS) return;
  int b = i / NCLS, c = i % NCLS;
  const float* fv = f1 + b * OFC1;
  const float* wv = w + c * OFC1;
  float s = bias[c];
  #pragma unroll 8
  for (int j = 0; j < OFC1; ++j) s += fv[j] * wv[j];
  out[i] = s;
}

extern "C" void kernel_launch(void* const* d_in, const int* in_sizes, int n_in,
                              void* d_out, int out_size, void* d_ws, size_t ws_size,
                              hipStream_t stream) {
  const float* x      = (const float*)d_in[0];
  const float* conv_w = (const float*)d_in[1];
  const float* conv_b = (const float*)d_in[2];
  const float* gcn_w  = (const float*)d_in[3];
  const float* gcn_b  = (const float*)d_in[4];
  const float* fc1_w  = (const float*)d_in[5];
  const float* fc1_b  = (const float*)d_in[6];
  const float* fc2_w  = (const float*)d_in[7];
  const float* fc2_b  = (const float*)d_in[8];

  // workspace layout (~71 MB total)
  char* p = (char*)d_ws;
  float*  pooled   = (float*)p;  p += (size_t)BATCH * CMID * PDIM * PDIM * 4;   // 16.8 MB
  float*  hfeat    = (float*)p;  p += (size_t)BATCH * NNODES * CGCN * 4;        // 33.6 MB
  __bf16* gbf      = (__bf16*)p; p += (size_t)BATCH * KFC1 * 2;                 // 16.8 MB
  float*  partials = (float*)p;  p += (size_t)SPLITS * BATCH * OFC1 * 4;        //  4.2 MB
  float*  f1       = (float*)p;  p += (size_t)BATCH * OFC1 * 4;                 //  16 KB

  conv_relu_pool <<<BATCH * CMID * 16, 256, 0, stream>>>(x, conv_w, conv_b, pooled);
  gcn_linear_wmma<<<(BATCH * NNODES) / 32, 256, 0, stream>>>(pooled, gcn_w, hfeat);
  gcn_agg        <<<(BATCH * NNODES * CGCN) / 256, 256, 0, stream>>>(hfeat, gcn_b, gbf);
  fc1_gemm_wmma  <<<SPLITS * 16, 32, 0, stream>>>(gbf, fc1_w, partials);
  fc1_reduce     <<<(BATCH * OFC1) / 256, 256, 0, stream>>>(partials, fc1_b, f1);
  fc2_kernel     <<<(BATCH * NCLS + 255) / 256, 256, 0, stream>>>(f1, fc2_w, fc2_b,
                                                                  (float*)d_out);
}
